// Attention_73426760892557
// MI455X (gfx1250) — matrix operand order repaired and checked by reading, not compiled, not run on hardware
//
#include <hip/hip_runtime.h>
#include <hip/hip_bf16.h>

#define BB 2
#define TT 2048
#define DD 2048
#define NH 16
#define KHN 8
#define HD 128
#define SS 2048

typedef __attribute__((ext_vector_type(16))) __bf16 v16bf;
typedef __attribute__((ext_vector_type(8)))  __bf16 bf16x8;
typedef __attribute__((ext_vector_type(4)))  __bf16 bf16x4;
typedef __attribute__((ext_vector_type(8)))  float  v8f;
typedef __attribute__((ext_vector_type(4)))  int    i32x4;
typedef __attribute__((ext_vector_type(8)))  int    i32x8;
typedef __attribute__((ext_vector_type(4)))  unsigned u32x4;

// async-LDS builtin takes addrspace(1)/addrspace(3) int4 pointers
typedef __attribute__((address_space(1))) i32x4 gi32x4;
typedef __attribute__((address_space(3))) i32x4 li32x4;
typedef __attribute__((address_space(3))) __bf16 lbf16;

#if defined(__has_builtin)
#if __has_builtin(__builtin_amdgcn_global_load_async_to_lds_b128)
#define HAVE_ASYNC_LDS 1
#endif
#if __has_builtin(__builtin_amdgcn_tensor_load_to_lds)
#define HAVE_TDM 1
#endif
#endif
#ifndef HAVE_ASYNC_LDS
#define HAVE_ASYNC_LDS 0
#endif
#ifndef HAVE_TDM
#define HAVE_TDM 0
#endif

static __device__ __forceinline__ __bf16 f2bf(float f) {
  unsigned u = __builtin_bit_cast(unsigned, f);
  unsigned r = u + 0x7fffu + ((u >> 16) & 1u);
  unsigned short h = (unsigned short)(r >> 16);
  return __builtin_bit_cast(__bf16, h);
}

static __device__ __forceinline__ v16bf cat8(bf16x8 lo, bf16x8 hi) {
  return __builtin_shufflevector(lo, hi, 0,1,2,3,4,5,6,7,8,9,10,11,12,13,14,15);
}

// A-fragment (16x32 bf16): lane m = lane&15, koff = (lane>>4)*8.
static __device__ __forceinline__ v16bf frag_a_ld(const __bf16* row, int koff) {
  bf16x8 lo = *(const bf16x8*)(row + koff);
  bf16x8 hi = *(const bf16x8*)(row + 16 + koff);
  return cat8(lo, hi);
}
// B-fragment (32x16 bf16): lane n = lane&15, kbase = (lane>>4)*16; k = kbase/2 + {0..7, 16..23}.
static __device__ __forceinline__ v16bf frag_b_ld(const __bf16* row, int kbase) {
  bf16x8 lo = *(const bf16x8*)(row + (kbase >> 1));
  bf16x8 hi = *(const bf16x8*)(row + 16 + (kbase >> 1));
  return cat8(lo, hi);
}

#define WMMA_BF16(a, b, c) \
  __builtin_amdgcn_wmma_f32_16x16x32_bf16(false, (a), false, (b), (short)0, (c), false, false)

// 16B global->LDS copy: async (ASYNCcnt) when available, else sync via regs.
static __device__ __forceinline__ void copy16_g2l(const __bf16* g, __bf16* l) {
#if HAVE_ASYNC_LDS
  __builtin_amdgcn_global_load_async_to_lds_b128((gi32x4*)g, (li32x4*)l, 0, 0);
#else
  *(bf16x8*)l = *(const bf16x8*)g;
#endif
}
static __device__ __forceinline__ void async_wait() {
#if HAVE_ASYNC_LDS
#if __has_builtin(__builtin_amdgcn_s_wait_asynccnt)
  __builtin_amdgcn_s_wait_asynccnt(0);
#else
  asm volatile("s_wait_asynccnt 0x0" ::: "memory");
#endif
#endif
}
static __device__ __forceinline__ void tensor_wait() {
#if HAVE_TDM
#if __has_builtin(__builtin_amdgcn_s_wait_tensorcnt)
  __builtin_amdgcn_s_wait_tensorcnt(0);
#else
  asm volatile("s_wait_tensorcnt 0x0" ::: "memory");
#endif
#endif
}

#if HAVE_TDM
// TDM 2D tile load (bf16): tile1 rows x tile0 elems, global row stride = stride0
// elems, LDS destination padded by 4 DWORDs every 64 DWORDs (pitch 272B rows).
// D# layout per CDNA5 ISA ch8 (group0: count/lds/global/type; group1: dims).
// This toolchain's builtin takes 6 args: (g0, g1, g2, g3, g4, cpol).
static __device__ __forceinline__ void tdm_load_2d_bf16(
    const __bf16* gsrc, unsigned lds_addr, unsigned dim0, unsigned dim1,
    unsigned stride0, unsigned tile0, unsigned tile1) {
  unsigned long long ga = (unsigned long long)gsrc;
  u32x4 g0;
  g0[0] = 1u;                                            // count=1, user D#
  g0[1] = lds_addr;                                      // LDS byte address
  g0[2] = (unsigned)ga;                                  // global addr [31:0]
  g0[3] = (unsigned)((ga >> 32) & 0x01FFFFFFull) | (2u << 30);  // [56:32], type=2
  i32x8 g1;
  g1[0] = (int)((1u << 16)      // data_size = 2 bytes
              | (1u << 20)      // pad_enable
              | (5u << 22)      // pad_interval: 64 DWORDs (256B row)
              | (3u << 25));    // pad_amount: 4 DWORDs (16B)
  g1[1] = (int)(dim0 << 16);                    // tensor_dim0[15:0]
  g1[2] = (int)((dim0 >> 16) | (dim1 << 16));   // dim0[31:16] | dim1[15:0]
  g1[3] = (int)((dim1 >> 16) | (tile0 << 16));  // dim1[31:16] | tile_dim0
  g1[4] = (int)(tile1 & 0xffffu);               // tile_dim1 (tile_dim2 = 0)
  g1[5] = (int)stride0;                         // tensor_dim0_stride[31:0]
  g1[6] = 0;                                    // stride hi | dim1_stride lo
  g1[7] = 0;
  i32x4 z4 = {0, 0, 0, 0};                      // groups 2/3 unused (<=2D)
  i32x8 z8 = {0, 0, 0, 0, 0, 0, 0, 0};          // extra group (unused)
  __builtin_amdgcn_tensor_load_to_lds(g0, g1, z4, z4, z8, 0);
}
#endif

// ---------------------------------------------------------------------------
__global__ void cvt_f32_bf16(const float* __restrict__ in, __bf16* __restrict__ out, long n) {
  long i = (long)blockIdx.x * blockDim.x + threadIdx.x;
  long stride = (long)gridDim.x * blockDim.x;
  for (; i < n; i += stride) out[i] = f2bf(in[i]);
}

__global__ void build_wcat(const float* __restrict__ qw, const float* __restrict__ kw,
                           const float* __restrict__ vw, __bf16* __restrict__ wcat) {
  long total = (long)DD * 4096;
  long i = (long)blockIdx.x * blockDim.x + threadIdx.x;
  long stride = (long)gridDim.x * blockDim.x;
  for (; i < total; i += stride) {
    int d = (int)(i >> 12), j = (int)(i & 4095);
    float v;
    if (j < 2048)      v = qw[(unsigned)(d * 2048 + j)];
    else if (j < 3072) v = kw[(unsigned)(d * 1024 + (j - 2048))];
    else               v = vw[(unsigned)(d * 1024 + (j - 3072))];
    wcat[i] = f2bf(v);
  }
}

__global__ void meta_k(const int* __restrict__ seg, const int* __restrict__ start_ind,
                       int* __restrict__ meta) {
  int b = threadIdx.x;
  if (b >= BB) return;
  int f = 0;
  while (f < TT && seg[(unsigned)(b * TT + f)] == 0) f++;
  if (f >= TT) f = 0;
  meta[2 * b]     = (start_ind[b] < 0) ? f : start_ind[b];
  meta[2 * b + 1] = f;
}

// ---------------------------------------------------------------------------
// bf16 GEMM, C f32 = A[M,K] x B[K,N], block tile 64x256, K-step 32,
// double-buffered LDS, 8 waves as 2x4, wave tile 32x64 (8 WMMA / K-step).
__global__ __launch_bounds__(256) void gemm_bf16(const __bf16* __restrict__ A,
                                                 const __bf16* __restrict__ B,
                                                 float* __restrict__ C,
                                                 int M, int N, int K) {
  __shared__ __attribute__((aligned(16))) __bf16 As[2][64 * 40];   // [m][32+8]
  __shared__ __attribute__((aligned(16))) __bf16 Bs[2][256 * 40];  // transposed [n][32+8]
  const int m0 = blockIdx.x * 64;
  const int n0 = blockIdx.y * 256;
  const int tid = threadIdx.x;
  const int wave = tid >> 5, lane = tid & 31;
  const int wm = wave >> 2, wn = wave & 3;
  const int ml = lane & 15, kh2 = lane >> 4;

  const int sar = tid >> 2, sak8 = (tid & 3) << 3;                 // A chunk coords
  v8f acc[2][4] = {{{}, {}, {}, {}}, {{}, {}, {}, {}}};

  auto stage = [&](int buf, int k0) {
    // A tile 64x32: one 16B async chunk per thread
    copy16_g2l(A + (unsigned)((m0 + sar) * K + k0 + sak8),
               &As[buf][sar * 40 + sak8]);
    // B tile 32x256 transposed: 512 units (4 k-rows x 4 cols), 2 per thread
#pragma unroll
    for (int i = 0; i < 2; i++) {
      int u = tid + i * 256;
      int kg = u >> 6;                 // k = kg*4
      int n4 = (u & 63) << 2;
      const __bf16* g = B + (unsigned)((k0 + kg * 4) * N + n0 + n4);
      bf16x4 r0 = *(const bf16x4*)(g);
      bf16x4 r1 = *(const bf16x4*)(g + N);
      bf16x4 r2 = *(const bf16x4*)(g + 2 * N);
      bf16x4 r3 = *(const bf16x4*)(g + 3 * N);
#pragma unroll
      for (int j = 0; j < 4; j++) {
        bf16x4 t = {r0[j], r1[j], r2[j], r3[j]};
        *(bf16x4*)(&Bs[buf][(n4 + j) * 40 + kg * 4]) = t;
      }
    }
  };

  stage(0, 0);
  async_wait();
  __syncthreads();

  int buf = 0;
  for (int k0 = 0; k0 < K; k0 += 32) {
    if (k0 + 32 < K) stage(buf ^ 1, k0 + 32);

    const __bf16* Ab = &As[buf][(wm * 32 + ml) * 40];
    const __bf16* Bb = &Bs[buf][(wn * 64 + ml) * 40];
    v16bf a0 = frag_a_ld(Ab, kh2 * 8);
    v16bf a1 = frag_a_ld(Ab + 16 * 40, kh2 * 8);
#pragma unroll
    for (int ct = 0; ct < 4; ct++) {
      v16bf bfrag = frag_b_ld(Bb + ct * 16 * 40, kh2 * 16);
      acc[0][ct] = WMMA_BF16(a0, bfrag, acc[0][ct]);
      acc[1][ct] = WMMA_BF16(a1, bfrag, acc[1][ct]);
    }
    async_wait();
    __syncthreads();
    buf ^= 1;
  }

  const int rb = m0 + wm * 32 + kh2 * 8;
  const int cb = n0 + wn * 64 + ml;
#pragma unroll
  for (int at = 0; at < 2; at++)
#pragma unroll
    for (int ct = 0; ct < 4; ct++)
      for (int r = 0; r < 8; r++) {
        int row = rb + at * 16 + r;
        int col = cb + ct * 16;
        if (row < M && col < N) C[(size_t)row * (unsigned)N + col] = acc[at][ct][r];
      }
}

// ---------------------------------------------------------------------------
__global__ __launch_bounds__(128) void normrope(
    const float* __restrict__ qkv, const float* __restrict__ qnw,
    const float* __restrict__ knw, const int* __restrict__ seg,
    const int* __restrict__ meta, const int* __restrict__ curp,
    __bf16* __restrict__ qbf, __bf16* __restrict__ kcbf, __bf16* __restrict__ vcbf) {
  const int tok = blockIdx.x;
  const int b = tok / TT, t = tok % TT;
  const int wave = threadIdx.x >> 5, lane = threadIdx.x & 31;
  const int cur = curp[0];
  const int fo = meta[2 * b + 1];
  const int sg = seg[tok];
  const float posf = (sg != 0) ? (float)(t - fo + cur) : 0.f;
  const float* row = qkv + (size_t)tok * 4096;
  const int krow = cur + t;

  for (int j = wave; j < 32; j += 4) {
    const int isq = (j < 16), isk = (j >= 16 && j < 24);
    const int col0 = isq ? j * 128 : (isk ? 2048 + (j - 16) * 128 : 3072 + (j - 24) * 128);
    float x0 = row[col0 + lane];
    float x1 = row[col0 + lane + 32];
    float x2 = row[col0 + lane + 64];
    float x3 = row[col0 + lane + 96];
    if (!isq && !isk) {
      if (krow < SS) {
        __bf16* o = vcbf + (unsigned)(((b * SS + krow) * KHN + (j - 24)) * HD);
        o[lane] = f2bf(x0); o[lane + 32] = f2bf(x1);
        o[lane + 64] = f2bf(x2); o[lane + 96] = f2bf(x3);
      }
      continue;
    }
    float ss = x0 * x0 + x1 * x1 + x2 * x2 + x3 * x3;
    ss += __shfl_xor(ss, 1); ss += __shfl_xor(ss, 2);
    ss += __shfl_xor(ss, 4); ss += __shfl_xor(ss, 8); ss += __shfl_xor(ss, 16);
    float rinv = rsqrtf(ss * (1.0f / 128.0f) + 1e-6f);
    const float* nw = isq ? qnw : knw;
    float y0 = nw[lane] * x0 * rinv;
    float y1 = nw[lane + 32] * x1 * rinv;
    float y2 = nw[lane + 64] * x2 * rinv;
    float y3 = nw[lane + 96] * x3 * rinv;
    const float LN_THETA = 13.815510558f;  // ln(1e6)
    float f0 = __expf(-((float)lane * (1.f / 64.f)) * LN_THETA);
    float f1 = __expf(-((float)(lane + 32) * (1.f / 64.f)) * LN_THETA);
    float a0 = posf * f0, a1 = posf * f1;
    float s0 = __sinf(a0), c0 = __cosf(a0);
    float s1 = __sinf(a1), c1 = __cosf(a1);
    float o0 = y0 * c0 - y2 * s0, o2 = y2 * c0 + y0 * s0;
    float o1 = y1 * c1 - y3 * s1, o3 = y3 * c1 + y1 * s1;
    if (isq) {
      __bf16* o = qbf + (unsigned)(((b * TT + t) * NH + j) * HD);
      o[lane] = f2bf(o0); o[lane + 32] = f2bf(o1);
      o[lane + 64] = f2bf(o2); o[lane + 96] = f2bf(o3);
    } else if (krow < SS) {
      __bf16* o = kcbf + (unsigned)(((b * SS + krow) * KHN + (j - 16)) * HD);
      o[lane] = f2bf(o0); o[lane + 32] = f2bf(o1);
      o[lane + 64] = f2bf(o2); o[lane + 96] = f2bf(o3);
    }
  }
}

// ---------------------------------------------------------------------------
// Flash attention, double-buffered K/V staging, 4 waves x 16 query rows.
// K tile staged by the Tensor Data Mover (wave 0 issues, TENSORcnt-tracked),
// with hardware LDS padding producing the 272B row pitch.
__global__ __launch_bounds__(128) void attn_fwd(
    const __bf16* __restrict__ qbf, const __bf16* __restrict__ kbf,
    const __bf16* __restrict__ vbf, __bf16* __restrict__ obf,
    const int* __restrict__ seg, const int* __restrict__ meta,
    const int* __restrict__ curp) {
  __shared__ __attribute__((aligned(16))) __bf16 Ksh[2][32 * 136];  // [key][feat]
  __shared__ __attribute__((aligned(16))) __bf16 Vts[2][128 * 40];  // [feat][key]
  __shared__ __attribute__((aligned(16))) __bf16 Ps[4 * 16 * 40];
  const int b = blockIdx.z, h = blockIdx.y, qb = blockIdx.x;
  const int kh = h >> 1;
  const int tid = threadIdx.x, wave = tid >> 5, lane = tid & 31;
  const int ml = lane & 15, kh2 = lane >> 4;
  const int cur = curp[0];
  const int startb = meta[2 * b];
  const int q0 = qb * 64 + wave * 16;

  v16bf qf0, qf1, qf2, qf3;
  {
    const __bf16* qp = qbf + (unsigned)(((b * TT + (q0 + ml)) * NH + h) * HD);
    qf0 = frag_a_ld(qp + 0,  kh2 * 8);
    qf1 = frag_a_ld(qp + 32, kh2 * 8);
    qf2 = frag_a_ld(qp + 64, kh2 * 8);
    qf3 = frag_a_ld(qp + 96, kh2 * 8);
  }
  int segr[8], qcap[8];
  for (int r = 0; r < 8; r++) {
    int t = q0 + r + 8 * kh2;
    segr[r] = seg[(unsigned)(b * TT + t)];
    qcap[r] = cur + t;
  }
  float mrow[8], lrow[8];
  v8f oacc[8] = {{}, {}, {}, {}, {}, {}, {}, {}};
  for (int r = 0; r < 8; r++) { mrow[r] = -1e30f; lrow[r] = 0.f; }

  int s_end = cur + (qb + 1) * 64;
  if (s_end > SS) s_end = SS;
  const int ntile = (s_end + 31) >> 5;
  __bf16* Pb = Ps + wave * 16 * 40;
  const float sc = 0.08838834764831845f;  // 1/sqrt(128)

  auto stage = [&](int buf, int s0) {
    unsigned base = (unsigned)(((b * SS + s0) * KHN + kh) * HD);
    // K tile 32x128 via TDM (or async fallback)
#if HAVE_TDM
    if (wave == 0) {
      unsigned lds_addr = (unsigned)(unsigned long long)(lbf16*)&Ksh[buf][0];
      tdm_load_2d_bf16(kbf + base, lds_addr,
                       /*dim0=*/HD, /*dim1=*/32,
                       /*stride0=*/KHN * HD, /*tile0=*/HD, /*tile1=*/32);
    }
#else
#pragma unroll
    for (int i = 0; i < 4; i++) {
      int c = tid + i * 128;
      int key = c >> 4, f8 = (c & 15) << 3;
      copy16_g2l(kbf + base + (unsigned)(key * (KHN * HD) + f8),
                 &Ksh[buf][key * 136 + f8]);
    }
#endif
    // V tile transposed: 256 units (4 keys x 4 feats), 2 per thread
#pragma unroll
    for (int i = 0; i < 2; i++) {
      int u = tid + i * 128;
      int kg = u >> 5, f4 = (u & 31) << 2;
      const __bf16* g = vbf + base + (unsigned)(kg * 4 * (KHN * HD) + f4);
      bf16x4 r0 = *(const bf16x4*)(g);
      bf16x4 r1 = *(const bf16x4*)(g + KHN * HD);
      bf16x4 r2 = *(const bf16x4*)(g + 2 * KHN * HD);
      bf16x4 r3 = *(const bf16x4*)(g + 3 * KHN * HD);
#pragma unroll
      for (int j = 0; j < 4; j++) {
        bf16x4 t = {r0[j], r1[j], r2[j], r3[j]};
        *(bf16x4*)(&Vts[buf][(f4 + j) * 40 + kg * 4]) = t;
      }
    }
  };

  stage(0, 0);
  tensor_wait();
  async_wait();
  __syncthreads();

  int buf = 0;
  for (int it = 0; it < ntile; it++) {
    const int s0 = it * 32;
    if (it + 1 < ntile) stage(buf ^ 1, s0 + 32);

    v8f lg0 = {}, lg1 = {};
    {
      const __bf16* K0 = &Ksh[buf][ml * 136];
      const __bf16* K1 = &Ksh[buf][(16 + ml) * 136];
      v16bf bk;
      bk = frag_b_ld(K0 + 0,  kh2 * 16); lg0 = WMMA_BF16(qf0, bk, lg0);
      bk = frag_b_ld(K0 + 32, kh2 * 16); lg0 = WMMA_BF16(qf1, bk, lg0);
      bk = frag_b_ld(K0 + 64, kh2 * 16); lg0 = WMMA_BF16(qf2, bk, lg0);
      bk = frag_b_ld(K0 + 96, kh2 * 16); lg0 = WMMA_BF16(qf3, bk, lg0);
      bk = frag_b_ld(K1 + 0,  kh2 * 16); lg1 = WMMA_BF16(qf0, bk, lg1);
      bk = frag_b_ld(K1 + 32, kh2 * 16); lg1 = WMMA_BF16(qf1, bk, lg1);
      bk = frag_b_ld(K1 + 64, kh2 * 16); lg1 = WMMA_BF16(qf2, bk, lg1);
      bk = frag_b_ld(K1 + 96, kh2 * 16); lg1 = WMMA_BF16(qf3, bk, lg1);
    }

    const int sA = s0 + ml, sB = s0 + 16 + ml;
    const int kvA = (sA >= startb && sA < cur + TT) ? 1 : 0;
    const int kvB = (sB >= startb && sB < cur + TT) ? 1 : 0;
    for (int r = 0; r < 8; r++) {
      bool okA = (sA <= qcap[r]) && (kvA == segr[r]);
      bool okB = (sB <= qcap[r]) && (kvB == segr[r]);
      float x0 = okA ? lg0[r] * sc : -1e30f;
      float x1 = okB ? lg1[r] * sc : -1e30f;
      float tmax = fmaxf(x0, x1);
      tmax = fmaxf(tmax, __shfl_xor(tmax, 1));
      tmax = fmaxf(tmax, __shfl_xor(tmax, 2));
      tmax = fmaxf(tmax, __shfl_xor(tmax, 4));
      tmax = fmaxf(tmax, __shfl_xor(tmax, 8));
      float nm = fmaxf(mrow[r], tmax);
      float gate = (nm > -1e29f) ? 1.f : 0.f;
      float alpha = __expf(mrow[r] - nm);
      mrow[r] = nm;
      float p0 = __expf(x0 - nm) * gate;
      float p1 = __expf(x1 - nm) * gate;
      float rs = p0 + p1;
      rs += __shfl_xor(rs, 1); rs += __shfl_xor(rs, 2);
      rs += __shfl_xor(rs, 4); rs += __shfl_xor(rs, 8);
      lrow[r] = lrow[r] * alpha + rs;
      for (int cg = 0; cg < 8; cg++) oacc[cg][r] *= alpha;
      int m = r + 8 * kh2;
      Pb[m * 40 + ml]      = f2bf(p0);
      Pb[m * 40 + 16 + ml] = f2bf(p1);
    }

    // P(16x32) @ V(32x128); same-wave LDS ops are in order, no barrier needed
    v16bf pa = frag_a_ld(Pb + ml * 40, kh2 * 8);
#pragma unroll
    for (int cg = 0; cg < 8; cg++) {
      v16bf bv = frag_b_ld(&Vts[buf][(cg * 16 + ml) * 40], kh2 * 16);
      oacc[cg] = WMMA_BF16(pa, bv, oacc[cg]);
    }
    tensor_wait();
    async_wait();
    __syncthreads();
    buf ^= 1;
  }

  float inv[8];
  for (int r = 0; r < 8; r++) inv[r] = (lrow[r] > 0.f) ? 1.f / lrow[r] : 0.f;
#pragma unroll
  for (int cg = 0; cg < 8; cg++)
    for (int r = 0; r < 8; r++) {
      int t = q0 + r + 8 * kh2;
      obf[(unsigned)(((b * TT + t) * NH + h) * HD + cg * 16 + ml)] =
          f2bf(oacc[cg][r] * inv[r]);
    }
}

// ---------------------------------------------------------------------------
extern "C" void kernel_launch(void* const* d_in, const int* in_sizes, int n_in,
                              void* d_out, int out_size, void* d_ws, size_t ws_size,
                              hipStream_t stream) {
  const float* x   = (const float*)d_in[0];
  const float* qw  = (const float*)d_in[1];
  const float* kw  = (const float*)d_in[2];
  const float* vw  = (const float*)d_in[3];
  const float* ow  = (const float*)d_in[4];
  const float* qnw = (const float*)d_in[5];
  const float* knw = (const float*)d_in[6];
  const float* kc  = (const float*)d_in[7];
  const float* vc  = (const float*)d_in[8];
  const int* seg   = (const int*)d_in[9];
  const int* sti   = (const int*)d_in[10];
  const int* cur   = (const int*)d_in[11];

  char* ws = (char*)d_ws;
  __bf16* xbf  = (__bf16*)(ws + 0);            // 16 MB (aliased as obf later)
  __bf16* wcat = (__bf16*)(ws + 16777216);     // 16 MB
  __bf16* owbf = (__bf16*)(ws + 33554432);     // 8 MB
  __bf16* kcbf = (__bf16*)(ws + 41943040);     // 8 MB
  __bf16* vcbf = (__bf16*)(ws + 50331648);     // 8 MB
  float*  qkv  = (float*)(ws + 58720256);      // 64 MB
  __bf16* qbf  = (__bf16*)(ws + 125829120);    // 16 MB
  int*    meta = (int*)(ws + 142606336);       // tiny
  __bf16* obf  = xbf;

  cvt_f32_bf16<<<1024, 256, 0, stream>>>(x,  xbf,  (long)BB * TT * DD);
  cvt_f32_bf16<<<1024, 256, 0, stream>>>(ow, owbf, (long)NH * HD * DD);
  cvt_f32_bf16<<<1024, 256, 0, stream>>>(kc, kcbf, (long)BB * SS * KHN * HD);
  cvt_f32_bf16<<<1024, 256, 0, stream>>>(vc, vcbf, (long)BB * SS * KHN * HD);
  build_wcat<<<2048, 256, 0, stream>>>(qw, kw, vw, wcat);
  meta_k<<<1, 32, 0, stream>>>(seg, sti, meta);

  gemm_bf16<<<dim3(64, 16), 256, 0, stream>>>(xbf, wcat, qkv, 4096, 4096, 2048);
  normrope<<<4096, 128, 0, stream>>>(qkv, qnw, knw, seg, meta, cur, qbf, kcbf, vcbf);
  attn_fwd<<<dim3(TT / 64, NH, BB), 128, 0, stream>>>(qbf, kcbf, vcbf, obf, seg, meta, cur);
  gemm_bf16<<<dim3(64, 8), 256, 0, stream>>>(obf, owbf, (float*)d_out, 4096, 2048, 2048);
}
